// MaskedMultiHeadAttention_42296837931767
// MI455X (gfx1250) — compile-verified, matrix operand order checked by reading, not run on hardware
//
#include <hip/hip_runtime.h>
#include <hip/hip_bf16.h>

// ---------------------------------------------------------------------------
// Causal MHA forward for MI455X (gfx1250, wave32).
// GEMMs: v_wmma_f32_16x16x32_f16 (f16 in, f32 accum). Softmax in f32.
// Attention stages K/V tiles into double-buffered LDS with
// global_load_async_to_lds_b128 (ASYNCcnt) shared by 4 waves per block.
// ---------------------------------------------------------------------------

typedef __attribute__((ext_vector_type(16))) _Float16 v16h;
typedef __attribute__((ext_vector_type(8)))  _Float16 v8h;
typedef __attribute__((ext_vector_type(8)))  float    v8f;

#define D_MODEL 1024
#define NH      16
#define DH      64
#define BATCH   2
#define SEQ     2048

// Workspace layout (bytes). Total ~42.4 MB.
constexpr size_t SZ_XH  = (size_t)BATCH * SEQ * D_MODEL * 2;   // 8 MB
constexpr size_t SZ_W   = (size_t)NH * DH * DH * 2;            // 128 KB
constexpr size_t SZ_WO  = (size_t)D_MODEL * D_MODEL * 2;       // 2 MB
constexpr size_t OFF_XH = 0;
constexpr size_t OFF_WQ = OFF_XH + SZ_XH;
constexpr size_t OFF_WK = OFF_WQ + SZ_W;
constexpr size_t OFF_WV = OFF_WK + SZ_W;
constexpr size_t OFF_WO = OFF_WV + SZ_W;
constexpr size_t OFF_Q  = OFF_WO + SZ_WO;
constexpr size_t OFF_K  = OFF_Q  + SZ_XH;
constexpr size_t OFF_VT = OFF_K  + SZ_XH;
constexpr size_t OFF_AO = OFF_VT + SZ_XH;

// ---------------------------------------------------------------------------
// Fragment loader for the 16-bit 16x32 A layout (also valid for B loaded as
// B^T rows): lane L -> row m = L%16, group g = L/16; halves j<8 hold
// K = 8g+j, halves j>=8 hold K = 16+8g+(j-8).  Two contiguous 16-byte runs.
// Works for global (global_load_b128) and LDS (ds_load_b128) pointers.
// ---------------------------------------------------------------------------
__device__ __forceinline__ v16h load_frag(const _Float16* base, int stride) {
  int lane = threadIdx.x & 31;
  int m = lane & 15;
  int g = lane >> 4;
  const _Float16* p = base + (size_t)m * stride + 8 * g;
  v16h f;
  reinterpret_cast<v8h*>(&f)[0] = *reinterpret_cast<const v8h*>(p);
  reinterpret_cast<v8h*>(&f)[1] = *reinterpret_cast<const v8h*>(p + 16);
  return f;
}

__device__ __forceinline__ v8f wmma_f16(v16h a, v16h b, v8f c) {
  // (neg_a, A, neg_b, B, c_mod, C, reuse_a, reuse_b)
  return __builtin_amdgcn_wmma_f32_16x16x32_f16(false, a, false, b, (short)0, c,
                                                false, false);
}

// ---------------------------------------------------------------------------
// CDNA5 async copy: LDS[ldsoff] <= 16B at global ptr, tracked by ASYNCcnt.
// Generic LDS pointers carry the wave-relative LDS byte offset in addr[31:0].
// ---------------------------------------------------------------------------
__device__ __forceinline__ uint32_t lds_byte_off(const void* p) {
  return (uint32_t)(uintptr_t)p;
}
__device__ __forceinline__ void async_copy16(uint32_t ldsoff, const _Float16* g) {
  asm volatile("global_load_async_to_lds_b128 %0, %1, off"
               :: "v"(ldsoff), "v"(g) : "memory");
}
__device__ __forceinline__ void wait_async_le4() {
  asm volatile("s_wait_asynccnt 0x4" ::: "memory");
}
__device__ __forceinline__ void wait_async_0() {
  asm volatile("s_wait_asynccnt 0x0" ::: "memory");
}

// ---------------------------------------------------------------------------
// Kernel 0: f32 -> f16 conversion (x, Wq, Wk, Wv, Wo)
// ---------------------------------------------------------------------------
__global__ void cvt_f32_f16_kernel(const float* __restrict__ in,
                                   _Float16* __restrict__ out, int n) {
  int i = blockIdx.x * blockDim.x + threadIdx.x;
  if (i < n) out[i] = (_Float16)in[i];
}

// ---------------------------------------------------------------------------
// Kernel 1: per-head QKV projection.  One wave per (b, h, 16-row s-tile).
// Q gets bias + 1/sqrt(64) pre-scale; V is stored transposed (B,H,DH,S).
// ---------------------------------------------------------------------------
__device__ __forceinline__ void proj_tiles(v16h a0, v16h a1,
                                           const _Float16* __restrict__ W,
                                           int h, v8f acc[4]) {
#pragma unroll
  for (int nt = 0; nt < 4; ++nt) {
    const _Float16* wb = W + (size_t)(h * DH + nt * 16) * DH;
    v16h b0 = load_frag(wb, DH);
    v16h b1 = load_frag(wb + 32, DH);
    v8f c = {};
    c = wmma_f16(a0, b0, c);
    c = wmma_f16(a1, b1, c);
    acc[nt] = c;
  }
}

__global__ void qkv_kernel(const _Float16* __restrict__ xh,
                           const _Float16* __restrict__ Wq, const float* __restrict__ bq,
                           const _Float16* __restrict__ Wk, const float* __restrict__ bk,
                           const _Float16* __restrict__ Wv, const float* __restrict__ bv,
                           _Float16* __restrict__ Q, _Float16* __restrict__ K,
                           _Float16* __restrict__ Vt) {
  int wave = blockIdx.x * (blockDim.x >> 5) + (threadIdx.x >> 5);
  int lane = threadIdx.x & 31;
  int st = wave % (SEQ / 16);
  int bh = wave / (SEQ / 16);
  int h  = bh % NH;
  int sbase = st * 16;
  int n = lane & 15, g = lane >> 4;

  const _Float16* xrow = xh + ((size_t)(bh / NH) * SEQ + sbase) * D_MODEL + h * DH;
  v16h a0 = load_frag(xrow, D_MODEL);
  v16h a1 = load_frag(xrow + 32, D_MODEL);

  // ---- Q: bias + 0.125 scale, row-major (B,H,S,DH) f16
  {
    v8f acc[4];
    proj_tiles(a0, a1, Wq, h, acc);
#pragma unroll
    for (int nt = 0; nt < 4; ++nt) {
      float bias = bq[h * DH + nt * 16 + n];
#pragma unroll
      for (int r = 0; r < 8; ++r) {
        float v = (acc[nt][r] + bias) * 0.125f;
        Q[((size_t)bh * SEQ + sbase + g * 8 + r) * DH + nt * 16 + n] = (_Float16)v;
      }
    }
  }
  // ---- K: bias only, row-major (B,H,S,DH) f16
  {
    v8f acc[4];
    proj_tiles(a0, a1, Wk, h, acc);
#pragma unroll
    for (int nt = 0; nt < 4; ++nt) {
      float bias = bk[h * DH + nt * 16 + n];
#pragma unroll
      for (int r = 0; r < 8; ++r) {
        float v = acc[nt][r] + bias;
        K[((size_t)bh * SEQ + sbase + g * 8 + r) * DH + nt * 16 + n] = (_Float16)v;
      }
    }
  }
  // ---- V: bias, stored transposed (B,H,DH,S); per-lane 16B contiguous store
  {
    v8f acc[4];
    proj_tiles(a0, a1, Wv, h, acc);
#pragma unroll
    for (int nt = 0; nt < 4; ++nt) {
      float bias = bv[h * DH + nt * 16 + n];
      v8h pack;
#pragma unroll
      for (int r = 0; r < 8; ++r) pack[r] = (_Float16)(acc[nt][r] + bias);
      *reinterpret_cast<v8h*>(Vt + ((size_t)bh * DH + nt * 16 + n) * SEQ +
                              sbase + g * 8) = pack;
    }
  }
}

// ---------------------------------------------------------------------------
// Kernel 2: causal flash attention.
// Block = 4 waves = 64 q-rows of one (b,h).  kv stepped by 32; K (32x64) and
// Vt slice (64x32) staged into double-buffered LDS with async-to-LDS copies,
// shared by all 4 waves.  Per-stage: 4 async b128 issues per lane.
// ---------------------------------------------------------------------------
__global__ void attn_kernel(const _Float16* __restrict__ Q,
                            const _Float16* __restrict__ K,
                            const _Float16* __restrict__ Vt,
                            _Float16* __restrict__ Aout) {
  __shared__ __align__(16) _Float16 sK[2][32 * DH];   // 2 x 4 KB
  __shared__ __align__(16) _Float16 sV[2][DH * 32];   // 2 x 4 KB, [d][token]
  __shared__ __align__(16) _Float16 sP[4][16 * 32];   // 4 x 1 KB

  const int tid   = threadIdx.x;
  const int wslot = tid >> 5;
  const int lane  = tid & 31;
  const int bq    = (blockIdx.x & 31) * 64;   // block q-row base
  const int bh    = blockIdx.x >> 5;
  const int h = bh % NH, b = bh / NH;
  const int qbase = bq + wslot * 16;
  const int n = lane & 15, g = lane >> 4;

  const _Float16* Kg = K  + (size_t)bh * SEQ * DH;
  const _Float16* Vg = Vt + (size_t)bh * DH * SEQ;

  // stage kv block [kvbase, kvbase+32) into buffer `buf` (whole block helps)
  auto stage = [&](int buf, int kvbase) {
    // K tile: 32 rows x 64 halves, fully contiguous 4 KB in global
    const _Float16* kg = Kg + (size_t)kvbase * DH;
    uint32_t ko = lds_byte_off(&sK[buf][0]);
#pragma unroll
    for (int i = 0; i < 2; ++i) {
      int c = tid + i * 128;                 // 16B chunk id, 0..255
      async_copy16(ko + c * 16, kg + c * 8);
    }
    // V tile: 64 rows (d) x 32 tokens; global row stride SEQ
    uint32_t vo = lds_byte_off(&sV[buf][0]);
#pragma unroll
    for (int i = 0; i < 2; ++i) {
      int c = tid + i * 128;
      int d = c >> 2, sub = c & 3;
      async_copy16(vo + (d * 32 + sub * 8) * 2,
                   Vg + (size_t)d * SEQ + kvbase + sub * 8);
    }
  };

  // Q fragments (global, once per wave)
  const _Float16* qrow = Q + ((size_t)bh * SEQ + qbase) * DH;
  v16h qa0 = load_frag(qrow, DH);
  v16h qa1 = load_frag(qrow + 32, DH);

  v8f acc[4];
#pragma unroll
  for (int nt = 0; nt < 4; ++nt) acc[nt] = v8f{};
  float rmax[8], rsum[8];
#pragma unroll
  for (int r = 0; r < 8; ++r) { rmax[r] = -1e30f; rsum[r] = 0.0f; }

  _Float16* P = sP[wslot];
  const int nb = bq / 32 + 2;   // kv blocks needed to cover rows bq..bq+63

  stage(0, 0);
  for (int jb = 0; jb < nb; ++jb) {
    const int kvbase = jb * 32;
    const int bufi = jb & 1;
    if (jb + 1 < nb) {
      stage(bufi ^ 1, kvbase + 32);
      wait_async_le4();            // stage(jb) done, stage(jb+1) in flight
    } else {
      wait_async_0();
    }
    __syncthreads();               // all waves' portions of buf[bufi] visible

    // ---- scores: two 16x16 tiles over K=64 (2 chunks), from LDS
    const _Float16* kb0 = &sK[bufi][0];
    v8f s0 = {}, s1 = {};
    {
      v16h t;
      t = load_frag(kb0, DH);                s0 = wmma_f16(qa0, t, s0);
      t = load_frag(kb0 + 32, DH);           s0 = wmma_f16(qa1, t, s0);
      t = load_frag(kb0 + 16 * DH, DH);      s1 = wmma_f16(qa0, t, s1);
      t = load_frag(kb0 + 16 * DH + 32, DH); s1 = wmma_f16(qa1, t, s1);
    }
    // ---- causal mask + online softmax (f32), stage P into LDS (f16)
    int col0 = kvbase + n, col1 = kvbase + 16 + n;
#pragma unroll
    for (int r = 0; r < 8; ++r) {
      int row = qbase + g * 8 + r;
      float x0 = (col0 <= row) ? s0[r] : -1e30f;
      float x1 = (col1 <= row) ? s1[r] : -1e30f;
      float mx = fmaxf(x0, x1);
#pragma unroll
      for (int off = 1; off < 16; off <<= 1)
        mx = fmaxf(mx, __shfl_xor(mx, off, 32));
      float nm = fmaxf(rmax[r], mx);
      float scale = __expf(rmax[r] - nm);
      float p0 = __expf(x0 - nm);
      float p1 = __expf(x1 - nm);
      float ps = p0 + p1;
#pragma unroll
      for (int off = 1; off < 16; off <<= 1)
        ps += __shfl_xor(ps, off, 32);
      rsum[r] = rsum[r] * scale + ps;
      rmax[r] = nm;
#pragma unroll
      for (int nt = 0; nt < 4; ++nt) acc[nt][r] *= scale;
      int rl = g * 8 + r;
      P[rl * 32 + n]      = (_Float16)p0;
      P[rl * 32 + 16 + n] = (_Float16)p1;
    }
    // ---- reload P as a 16x32 A fragment (same-wave LDS RAW, in order)
    v16h pf;
    {
      int m = lane & 15;
      const _Float16* pp = P + m * 32 + 8 * g;
      reinterpret_cast<v8h*>(&pf)[0] = *reinterpret_cast<const v8h*>(pp);
      reinterpret_cast<v8h*>(&pf)[1] = *reinterpret_cast<const v8h*>(pp + 16);
    }
    // ---- O += P @ V  (V tile in LDS, [d][token] -> contiguous B fragments)
#pragma unroll
    for (int nt = 0; nt < 4; ++nt) {
      v16h bf = load_frag(&sV[bufi][nt * 16 * 32], 32);
      acc[nt] = wmma_f16(pf, bf, acc[nt]);
    }
    __syncthreads();               // everyone done reading buf[bufi]
  }

  // ---- normalize, write attn output (B,S,D_MODEL) f16, head-major concat
#pragma unroll
  for (int r = 0; r < 8; ++r) {
    float inv = 1.0f / rsum[r];
    size_t row = (size_t)b * SEQ + qbase + g * 8 + r;
#pragma unroll
    for (int nt = 0; nt < 4; ++nt)
      Aout[row * D_MODEL + h * DH + nt * 16 + n] =
          (_Float16)(acc[nt][r] * inv);
  }
}

// ---------------------------------------------------------------------------
// Kernel 3: output projection  out = A @ Wo^T + bo  (f32 output).
// One wave per 16x64 tile (A fragment shared by 4 accumulators).
// ---------------------------------------------------------------------------
__global__ void out_proj_kernel(const _Float16* __restrict__ A,
                                const _Float16* __restrict__ Wo,
                                const float* __restrict__ bo,
                                float* __restrict__ out) {
  int wave = blockIdx.x * 4 + (threadIdx.x >> 5);
  int lane = threadIdx.x & 31;
  int ct = wave % (D_MODEL / DH);          // 16 col groups of 64
  int rt = wave / (D_MODEL / DH);          // 256 row tiles
  int rbase = rt * 16, cbase = ct * DH;
  int n = lane & 15, g = lane >> 4;

  const _Float16* arow = A  + (size_t)rbase * D_MODEL;
  const _Float16* wrow = Wo + (size_t)cbase * D_MODEL;
  v8f acc[4];
#pragma unroll
  for (int nt = 0; nt < 4; ++nt) acc[nt] = v8f{};

#pragma unroll 2
  for (int kc = 0; kc < D_MODEL; kc += 32) {
    if (kc + 128 < D_MODEL) {
      __builtin_prefetch(arow + kc + 128, 0, 1);
      __builtin_prefetch(wrow + kc + 128, 0, 1);
    }
    v16h af = load_frag(arow + kc, D_MODEL);
#pragma unroll
    for (int nt = 0; nt < 4; ++nt) {
      v16h bf = load_frag(wrow + (size_t)nt * 16 * D_MODEL + kc, D_MODEL);
      acc[nt] = wmma_f16(af, bf, acc[nt]);
    }
  }
#pragma unroll
  for (int nt = 0; nt < 4; ++nt) {
    float bias = bo[cbase + nt * 16 + n];
#pragma unroll
    for (int r = 0; r < 8; ++r)
      out[(size_t)(rbase + g * 8 + r) * D_MODEL + cbase + nt * 16 + n] =
          acc[nt][r] + bias;
  }
}

// ---------------------------------------------------------------------------
extern "C" void kernel_launch(void* const* d_in, const int* in_sizes, int n_in,
                              void* d_out, int out_size, void* d_ws, size_t ws_size,
                              hipStream_t stream) {
  const float* x  = (const float*)d_in[0];
  const float* Wq = (const float*)d_in[1];
  const float* bq = (const float*)d_in[2];
  const float* Wk = (const float*)d_in[3];
  const float* bk = (const float*)d_in[4];
  const float* Wv = (const float*)d_in[5];
  const float* bv = (const float*)d_in[6];
  const float* Wo = (const float*)d_in[7];
  const float* bo = (const float*)d_in[8];
  float* out = (float*)d_out;

  char* ws = (char*)d_ws;
  _Float16* xh16 = (_Float16*)(ws + OFF_XH);
  _Float16* Wq16 = (_Float16*)(ws + OFF_WQ);
  _Float16* Wk16 = (_Float16*)(ws + OFF_WK);
  _Float16* Wv16 = (_Float16*)(ws + OFF_WV);
  _Float16* Wo16 = (_Float16*)(ws + OFF_WO);
  _Float16* Q16  = (_Float16*)(ws + OFF_Q);
  _Float16* K16  = (_Float16*)(ws + OFF_K);
  _Float16* Vt16 = (_Float16*)(ws + OFF_VT);
  _Float16* AO16 = (_Float16*)(ws + OFF_AO);

  const int nx  = BATCH * SEQ * D_MODEL;   // 4,194,304
  const int nw  = NH * DH * DH;            // 65,536
  const int nwo = D_MODEL * D_MODEL;       // 1,048,576
  cvt_f32_f16_kernel<<<(nx  + 255) / 256, 256, 0, stream>>>(x,  xh16, nx);
  cvt_f32_f16_kernel<<<(nw  + 255) / 256, 256, 0, stream>>>(Wq, Wq16, nw);
  cvt_f32_f16_kernel<<<(nw  + 255) / 256, 256, 0, stream>>>(Wk, Wk16, nw);
  cvt_f32_f16_kernel<<<(nw  + 255) / 256, 256, 0, stream>>>(Wv, Wv16, nw);
  cvt_f32_f16_kernel<<<(nwo + 255) / 256, 256, 0, stream>>>(Wo, Wo16, nwo);

  // B*NH*(SEQ/16) = 4096 waves -> 1024 blocks of 4 waves
  qkv_kernel<<<1024, 128, 0, stream>>>(xh16, Wq16, bq, Wk16, bk, Wv16, bv,
                                       Q16, K16, Vt16);
  // B*NH*(SEQ/64) = 1024 blocks, 4 waves sharing LDS K/V tiles
  attn_kernel<<<1024, 128, 0, stream>>>(Q16, K16, Vt16, AO16);
  // 256 row tiles x 16 col groups = 4096 waves -> 1024 blocks
  out_proj_kernel<<<1024, 128, 0, stream>>>(AO16, Wo16, bo, out);
}
